// MultiHeadAttention_75849122447478
// MI455X (gfx1250) — matrix operand order repaired and checked by reading, not compiled, not run on hardware
//
#include <hip/hip_runtime.h>
#include <hip/hip_bf16.h>

// MI455X (gfx1250) fused LoRA-MHA.
// Compute-bound (~206 GFLOP vs ~100MB traffic @23.3TB/s): all matmuls via
// v_wmma_f32_16x16x32_bf16 (wave32). LoRA folded into effective weights.
// WMMA A-fragments loaded as contiguous b128 pairs, B-frags per element; K is stored
// pre-transposed so score B-frag lanes read consecutive keys; 1/sqrt(dh) folded into the
// RoPE'd Q (exact, power of two); causal masking only in the single diagonal
// k-block. Workspace use: 88 MB (all L2-resident).

#define Bsz 4
#define Tsz 2048
#define Dsz 1024
#define Hsz 16
#define DH  64
#define Rsz 8
#define LORA_SCALE 4.0f

typedef __bf16 bf16_t;
typedef bf16_t v16bf __attribute__((ext_vector_type(16)));
typedef float  v8f   __attribute__((ext_vector_type(8)));

union Frag16 { v16bf v; unsigned short u[16]; uint4 q[2]; };
union Frag32 { v8f   v; float f[8]; };

__device__ __forceinline__ unsigned short f2bf(float f) {
  union { float f; unsigned int u; } v; v.f = f;
  unsigned int r = v.u + 0x7FFFu + ((v.u >> 16) & 1u);   // round-nearest-even
  return (unsigned short)(r >> 16);
}
__device__ __forceinline__ float bf2f(unsigned short h) {
  union { unsigned int u; float f; } v; v.u = ((unsigned int)h) << 16; return v.f;
}

// 16-bit 16x32 A-fragment (ISA 7.12.2): lane (row, half hh) holds
// K = hh*8..hh*8+7 (elements 0..7) and K = 16+hh*8..+7 (elements 8..15),
// i.e. two contiguous 16B runs -> two b128 loads.
__device__ __forceinline__ Frag16 load_a16(const unsigned short* p, int hh) {
  Frag16 f;
  const uint4* q = (const uint4*)(p + hh * 8);
  f.q[0] = q[0];
  f.q[1] = q[2];
  return f;
}

// ---------------- weight fold (+LoRA) & convert to bf16 ----------------
__global__ void fold_weight_kernel(const float* __restrict__ W,
                                   const float* __restrict__ Am,
                                   const float* __restrict__ Bm,
                                   unsigned short* __restrict__ Wout,
                                   int useLora) {
  int idx = blockIdx.x * blockDim.x + threadIdx.x;
  if (idx >= Dsz * Dsz) return;
  int d1 = idx >> 10, d2 = idx & (Dsz - 1);
  float acc = W[idx];
  if (useLora) {
    float s = 0.f;
#pragma unroll
    for (int r = 0; r < Rsz; ++r) s += Am[d1 * Rsz + r] * Bm[r * Dsz + d2];
    acc += LORA_SCALE * s;
  }
  Wout[idx] = f2bf(acc);
}

__global__ void convert_kernel(const float* __restrict__ in,
                               unsigned short* __restrict__ out, int n) {
  int idx = blockIdx.x * blockDim.x + threadIdx.x;
  if (idx < n) out[idx] = f2bf(in[idx]);
}

// ---------------- generic bf16 WMMA GEMM: C[M,N] = A[M,K] @ B[K,N] --------
__device__ __forceinline__ void store_out(float* p, float v) { *p = v; }
__device__ __forceinline__ void store_out(unsigned short* p, float v) { *p = f2bf(v); }

template <typename OutT>
__global__ __launch_bounds__(128) void gemm_bf16_kernel(
    const unsigned short* __restrict__ A, const unsigned short* __restrict__ Bw,
    OutT* __restrict__ C, int M, int N, int K) {
  const int lane = threadIdx.x & 31;
  const int wave = threadIdx.x >> 5;
  const int tilesN = N >> 6;
  const int tm = blockIdx.x / tilesN;
  const int tn = blockIdx.x % tilesN;
  const int rowbase = tm * 64 + wave * 16;
  const int colbase = tn * 64;
  const int m16 = lane & 15, hh = lane >> 4;

  Frag32 acc[4];
#pragma unroll
  for (int n = 0; n < 4; ++n)
#pragma unroll
    for (int r = 0; r < 8; ++r) acc[n].f[r] = 0.f;

  const unsigned short* arow = A + (size_t)(rowbase + m16) * K;
  for (int kb = 0; kb < K; kb += 32) {
    if (kb + 32 < K) {  // global_prefetch_b8 for the next k-slab
      __builtin_prefetch(arow + kb + 32, 0, 0);
      __builtin_prefetch(Bw + (size_t)(kb + 32 + lane) * N + colbase, 0, 0);
    }
    Frag16 af = load_a16(arow + kb, hh);
#pragma unroll
    for (int n = 0; n < 4; ++n) {
      Frag16 bfg;
#pragma unroll
      for (int s = 0; s < 16; ++s)
        bfg.u[s] = Bw[(size_t)(kb + (s < 8 ? hh * 8 + s : 16 + hh * 8 + s - 8)) * N + colbase + n * 16 + m16];
      acc[n].v = __builtin_amdgcn_wmma_f32_16x16x32_bf16(
          false, af.v, false, bfg.v, (short)0, acc[n].v, false, false);
    }
  }
#pragma unroll
  for (int n = 0; n < 4; ++n)
#pragma unroll
    for (int r = 0; r < 8; ++r)
      store_out(&C[(size_t)(rowbase + r + 8 * hh) * N + colbase + n * 16 + m16],
                acc[n].f[r]);
}

// ---------------- RoPE ----------------
// In-place on Q bf16 [B,T,D]; folds the 1/sqrt(dh)=0.125 score scale (exact).
__global__ void rope_q_kernel(unsigned short* __restrict__ X) {
  int idx = blockIdx.x * blockDim.x + threadIdx.x;  // pair index
  if (idx >= Bsz * Tsz * Dsz / 2) return;
  int pc = idx & (Dsz / 2 - 1);
  int bt = idx >> 9;              // D/2 = 512
  int t  = bt & (Tsz - 1);
  int col = pc * 2;
  int f = (col & (DH - 1)) >> 1;  // 0..31
  float theta = (float)t * __powf(10000.0f, -(float)f / 32.0f);
  float s, c;
  __sincosf(theta, &s, &c);
  unsigned short* p = X + (size_t)bt * Dsz + col;
  float e = bf2f(p[0]), o = bf2f(p[1]);
  p[0] = f2bf((e * c - o * s) * 0.125f);
  p[1] = f2bf((e * s + o * c) * 0.125f);
}

// K: read [B,T,D], apply RoPE, write transposed KT[b,h,dh,T] so that score
// B-fragments (lane=key, elements=dims, stride T) read 16 consecutive keys per element.
__global__ void rope_kt_kernel(const unsigned short* __restrict__ Kin,
                               unsigned short* __restrict__ KT) {
  int idx = blockIdx.x * blockDim.x + threadIdx.x;  // pair index
  if (idx >= Bsz * Tsz * Dsz / 2) return;
  int pc = idx & (Dsz / 2 - 1);
  int bt = idx >> 9;
  int b  = bt >> 11;              // T = 2048
  int t  = bt & (Tsz - 1);
  int col = pc * 2;
  int h = col >> 6;
  int d = col & (DH - 1);
  int f = d >> 1;
  float theta = (float)t * __powf(10000.0f, -(float)f / 32.0f);
  float s, c;
  __sincosf(theta, &s, &c);
  const unsigned short* p = Kin + (size_t)bt * Dsz + col;
  float e = bf2f(p[0]), o = bf2f(p[1]);
  unsigned short* q = KT + ((size_t)(b * Hsz + h) * DH + d) * Tsz + t;
  q[0]   = f2bf(e * c - o * s);
  q[Tsz] = f2bf(e * s + o * c);   // dim d+1, stride T
}

// ---------------- flash attention: 1 wave = 16 q-rows of one (b,h) --------
// One 32-key step of online-softmax attention. MASK=true only for the single
// diagonal block; full blocks carry no compare/select code.
template <bool MASK>
__device__ __forceinline__ void flash_step(
    int kb, int qbase, const unsigned short* __restrict__ kt,
    const unsigned short* __restrict__ vrow, unsigned short* pl,
    const Frag16* aq, Frag32* oacc, float* mrow, float* lrow,
    int lane, int m16, int hh) {
  // S = Q @ K^T : two 16x16 score tiles (key chunks), K=32 x2 over dh.
  Frag32 sc[2];
#pragma unroll
  for (int c = 0; c < 2; ++c) {
#pragma unroll
    for (int r = 0; r < 8; ++r) sc[c].f[r] = 0.f;
#pragma unroll
    for (int d = 0; d < 2; ++d) {
      Frag16 kfg;
#pragma unroll
      for (int s = 0; s < 16; ++s)
        kfg.u[s] = kt[(size_t)(d * 32 + (s < 8 ? hh * 8 + s : 16 + hh * 8 + s - 8)) * Tsz + kb + c * 16 + m16];
      sc[c].v = __builtin_amdgcn_wmma_f32_16x16x32_bf16(
          false, aq[d].v, false, kfg.v, (short)0, sc[c].v, false, false);
    }
  }
  // online softmax; C-layout row r+8*hh aligns with shuffle halves
#pragma unroll
  for (int r = 0; r < 8; ++r) {
    float s0 = sc[0].f[r];   // already scaled by 1/sqrt(dh) via Q
    float s1 = sc[1].f[r];
    if (MASK) {
      int row = qbase + r + 8 * hh;
      if (kb + m16 > row)      s0 = -1e30f;   // causal
      if (kb + 16 + m16 > row) s1 = -1e30f;
    }
    float mx = fmaxf(s0, s1);
#pragma unroll
    for (int d = 1; d < 16; d <<= 1) mx = fmaxf(mx, __shfl_xor(mx, d, 32));
    float mnew = fmaxf(mrow[r], mx);
    float p0 = __expf(s0 - mnew), p1 = __expf(s1 - mnew);
    float ps = p0 + p1;
#pragma unroll
    for (int d = 1; d < 16; d <<= 1) ps += __shfl_xor(ps, d, 32);
    float corr = __expf(mrow[r] - mnew);
    lrow[r] = lrow[r] * corr + ps;
    mrow[r] = mnew;
#pragma unroll
    for (int n = 0; n < 4; ++n) oacc[n].f[r] *= corr;
    pl[(r + 8 * hh) * 32 + m16]      = f2bf(p0);
    pl[(r + 8 * hh) * 32 + 16 + m16] = f2bf(p1);
  }
  asm volatile("s_wait_dscnt 0" ::: "memory");  // intra-wave P store->load
  __builtin_amdgcn_wave_barrier();
  // P A-frag (16x32 over the 32 keys) from LDS: contiguous b128 pair
  Frag16 pa;
  {
    const uint4* pp = (const uint4*)(pl + m16 * 32 + hh * 8);
    pa.q[0] = pp[0];
    pa.q[1] = pp[2];
  }
  // O += P @ V : V B-frag lane = dim column, elements = keys in table order
#pragma unroll
  for (int n = 0; n < 4; ++n) {
    Frag16 vfg;
#pragma unroll
    for (int s = 0; s < 16; ++s)
      vfg.u[s] = vrow[(size_t)(kb + (s < 8 ? hh * 8 + s : 16 + hh * 8 + s - 8)) * Dsz + n * 16 + m16];
    oacc[n].v = __builtin_amdgcn_wmma_f32_16x16x32_bf16(
        false, pa.v, false, vfg.v, (short)0, oacc[n].v, false, false);
  }
}

__global__ __launch_bounds__(128) void flash_kernel(
    const unsigned short* __restrict__ Q, const unsigned short* __restrict__ KT,
    const unsigned short* __restrict__ Vm, unsigned short* __restrict__ O) {
  __shared__ unsigned short Plds[4][16 * 32];  // per-wave P transpose tile
  const int lane = threadIdx.x & 31;
  const int wave = threadIdx.x >> 5;
  const int tile = blockIdx.x * 4 + wave;
  const int QT = Tsz / 16;
  const int qt = tile % QT;
  const int h  = (tile / QT) % Hsz;
  const int b  = tile / (QT * Hsz);
  const int qbase = qt * 16;
  const size_t base = (size_t)b * Tsz * Dsz;
  const int hcol = h * DH;
  const int m16 = lane & 15, hh = lane >> 4;
  const unsigned short* kt = KT + (size_t)(b * Hsz + h) * DH * Tsz;
  const unsigned short* vrow = Vm + base + hcol;
  unsigned short* pl = &Plds[wave][0];

  // Q A-fragments (two K=32 chunks over dh=64): contiguous b128 pairs
  Frag16 aq[2];
  {
    const unsigned short* qrow = Q + base + (size_t)(qbase + m16) * Dsz + hcol;
    aq[0] = load_a16(qrow, hh);
    aq[1] = load_a16(qrow + 32, hh);
  }

  Frag32 oacc[4];
  float mrow[8], lrow[8];
#pragma unroll
  for (int r = 0; r < 8; ++r) { mrow[r] = -1e30f; lrow[r] = 0.f; }
#pragma unroll
  for (int n = 0; n < 4; ++n)
#pragma unroll
    for (int r = 0; r < 8; ++r) oacc[n].f[r] = 0.f;

  // nfull unmasked 32-key blocks, then exactly one diagonal (masked) block.
  const int nfull = qbase >> 5;
  int kb = 0;
  for (int i = 0; i < nfull; ++i, kb += 32)
    flash_step<false>(kb, qbase, kt, vrow, pl, aq, oacc, mrow, lrow, lane, m16, hh);
  flash_step<true>(kb, qbase, kt, vrow, pl, aq, oacc, mrow, lrow, lane, m16, hh);

  // normalize and store bf16 into [B,T,D] (head-interleaved columns)
#pragma unroll
  for (int n = 0; n < 4; ++n)
#pragma unroll
    for (int r = 0; r < 8; ++r) {
      float val = oacc[n].f[r] / lrow[r];
      O[base + (size_t)(qbase + r + 8 * hh) * Dsz + hcol + n * 16 + m16] =
          f2bf(val);
    }
}

// ---------------- host launch ----------------
extern "C" void kernel_launch(void* const* d_in, const int* in_sizes, int n_in,
                              void* d_out, int out_size, void* d_ws, size_t ws_size,
                              hipStream_t stream) {
  (void)in_sizes; (void)n_in; (void)out_size; (void)ws_size;
  const float* query = (const float*)d_in[0];
  const float* wq = (const float*)d_in[1];
  const float* wk = (const float*)d_in[2];
  const float* wv = (const float*)d_in[3];
  const float* wo = (const float*)d_in[4];
  const float* Aq = (const float*)d_in[5];
  const float* Bq = (const float*)d_in[6];
  const float* Av = (const float*)d_in[7];
  const float* Bv = (const float*)d_in[8];

  unsigned short* ws = (unsigned short*)d_ws;
  const size_t WW = (size_t)Dsz * Dsz;        // 1M elems
  const size_t XX = (size_t)Bsz * Tsz * Dsz;  // 8M elems
  unsigned short* wqb  = ws;
  unsigned short* wkb  = wqb + WW;
  unsigned short* wvb  = wkb + WW;
  unsigned short* wob  = wvb + WW;
  unsigned short* qx   = wob + WW;   // bf16 query
  unsigned short* qb   = qx + XX;
  unsigned short* kbuf = qb + XX;
  unsigned short* vbuf = kbuf + XX;
  unsigned short* ktb  = vbuf + XX;  // K^T [B,H,DH,T]
  unsigned short* attn = qx;         // alias: qx dead after QKV GEMMs
  // total ws: (4*WW + 5*XX)*2B = 88 MB

  const int M = Bsz * Tsz;  // 8192
  const int wgrid = (Dsz * Dsz + 255) / 256;
  fold_weight_kernel<<<wgrid, 256, 0, stream>>>(wq, Aq, Bq, wqb, 1);
  fold_weight_kernel<<<wgrid, 256, 0, stream>>>(wk, nullptr, nullptr, wkb, 0);
  fold_weight_kernel<<<wgrid, 256, 0, stream>>>(wv, Av, Bv, wvb, 1);
  fold_weight_kernel<<<wgrid, 256, 0, stream>>>(wo, nullptr, nullptr, wob, 0);
  convert_kernel<<<(int)((XX + 255) / 256), 256, 0, stream>>>(query, qx, (int)XX);

  const int gblocks = (M / 64) * (Dsz / 64);  // 2048
  gemm_bf16_kernel<unsigned short><<<gblocks, 128, 0, stream>>>(qx, wqb, qb, M, Dsz, Dsz);
  gemm_bf16_kernel<unsigned short><<<gblocks, 128, 0, stream>>>(qx, wkb, kbuf, M, Dsz, Dsz);
  gemm_bf16_kernel<unsigned short><<<gblocks, 128, 0, stream>>>(qx, wvb, vbuf, M, Dsz, Dsz);

  const int rgrid = (int)((XX / 2 + 255) / 256);
  rope_q_kernel<<<rgrid, 256, 0, stream>>>(qb);
  rope_kt_kernel<<<rgrid, 256, 0, stream>>>(kbuf, ktb);

  flash_kernel<<<(Bsz * Hsz * (Tsz / 16)) / 4, 128, 0, stream>>>(qb, ktb, vbuf, attn);

  gemm_bf16_kernel<float><<<gblocks, 128, 0, stream>>>(attn, wob, (float*)d_out, M, Dsz, Dsz);
}
